// Attention_27041114096339
// MI455X (gfx1250) — matrix operand order repaired and checked by reading, not compiled, not run on hardware
//
#include <hip/hip_runtime.h>

typedef __attribute__((ext_vector_type(16))) __bf16 v16bf;
typedef __attribute__((ext_vector_type(8)))  __bf16 v8bf;
typedef __attribute__((ext_vector_type(8)))  float  v8f;

#define B_  64
#define T_  1024
#define E_  512
#define A_  128
#define F_  32

// ---------------------------------------------------------------------------
// Pack Wk (E x A, f32 row-major) into bf16 WMMA B-fragment order:
//   [eb(16)][nb(8)][lane(32)][elem(16)]
//   lane: n = nb*16 + (lane&15), k = (lane<16 ? 0 : 16) + elem  (K=32 slice eb)
// ---------------------------------------------------------------------------
__global__ __launch_bounds__(256) void k_packWk(const float* __restrict__ Wk,
                                                __bf16* __restrict__ wkp) {
    int gid = blockIdx.x * 256 + threadIdx.x;      // 65536 total
    int e   = gid & 15;
    int fl  = (gid >> 4) & 31;
    int nb  = (gid >> 9) & 7;
    int eb  = gid >> 12;
    int n   = nb * 16 + (fl & 15);
    int k   = ((fl & 16) ? 16 : 0) + e;
    wkp[gid] = (__bf16)Wk[(size_t)(eb * 32 + k) * A_ + n];
}

// ---------------------------------------------------------------------------
// q = query @ Wq   (B x R) @ (R x A) -> (B x A)
// ---------------------------------------------------------------------------
__global__ __launch_bounds__(128) void k_qproj(const float* __restrict__ query,
                                               const float* __restrict__ Wq,
                                               float* __restrict__ qbuf) {
    int b = blockIdx.x, a = threadIdx.x;
    const float* qr = query + (size_t)b * 1024;
    float acc = 0.f;
    #pragma unroll 4
    for (int r = 0; r < 1024; ++r) acc += qr[r] * Wq[(size_t)r * A_ + a];
    qbuf[b * A_ + a] = acc;
}

// ---------------------------------------------------------------------------
// locq[b,t,:] = q[b,:] + BN(relu(conv1d(aw_cat)))[b,:,t] @ Wloc
// one block = (b, 128-t chunk), 128 threads (one t each)
// ---------------------------------------------------------------------------
__global__ __launch_bounds__(128) void k_loc(const float* __restrict__ awcat,
                                             const float* __restrict__ convw,
                                             const float* __restrict__ gamma,
                                             const float* __restrict__ beta,
                                             const float* __restrict__ mean,
                                             const float* __restrict__ var,
                                             const float* __restrict__ Wloc,
                                             const float* __restrict__ qbuf,
                                             float* __restrict__ locq) {
    __shared__ float awsh[2][158];      // 128 + 30 halo
    __shared__ float cwsh[F_ * 62];     // conv weights (F,2,31)
    __shared__ float wlsh[F_ * A_];     // Wloc
    __shared__ float scsh[F_], bish[F_];

    int b  = blockIdx.x >> 3;
    int t0 = (blockIdx.x & 7) * 128;
    int tid = threadIdx.x;

    for (int i = tid; i < 316; i += 128) {
        int row = i / 158, col = i % 158;
        int g = t0 + col - 15;
        awsh[row][col] = (g >= 0 && g < T_) ? awcat[((size_t)b * 2 + row) * T_ + g] : 0.f;
    }
    for (int i = tid; i < F_ * 62; i += 128) cwsh[i] = convw[i];
    for (int i = tid; i < F_ * A_; i += 128) wlsh[i] = Wloc[i];
    if (tid < F_) {
        float sc = rsqrtf(var[tid] + 1e-5f) * gamma[tid];
        scsh[tid] = sc;
        bish[tid] = beta[tid] - mean[tid] * sc;
    }
    __syncthreads();

    float fv[F_];
    #pragma unroll 2
    for (int f = 0; f < F_; ++f) {
        float acc = 0.f;
        #pragma unroll
        for (int cch = 0; cch < 2; ++cch) {
            const float* wp = &cwsh[f * 62 + cch * 31];
            const float* ap = &awsh[cch][tid];
            #pragma unroll
            for (int dk = 0; dk < 31; ++dk) acc += ap[dk] * wp[dk];
        }
        acc = fmaxf(acc, 0.f);                       // relu then BN (eval)
        fv[f] = acc * scsh[f] + bish[f];
    }

    const float* qb = qbuf + (size_t)b * A_;
    float* op = locq + ((size_t)b * T_ + t0 + tid) * A_;
    for (int a = 0; a < A_; ++a) {
        float s = qb[a];
        #pragma unroll
        for (int f = 0; f < F_; ++f) s += fv[f] * wlsh[f * A_ + a];
        op[a] = s;
    }
}

// ---------------------------------------------------------------------------
// Stage four 16x32 key tiles (64 t-rows x one K=32 slice) into LDS in WMMA
// A-fragment order, f32 -> bf16. Each thread: 8 contiguous floats (2x b128
// load), one 16B ds_store.
// ---------------------------------------------------------------------------
__device__ __forceinline__ void stage_tiles(const float* __restrict__ keyp,
                                            __bf16* __restrict__ dst,
                                            int eb, int tid) {
    int idx0  = tid * 8;
    int tile  = idx0 >> 9;           // 0..3
    int idx9  = idx0 & 511;
    int fl    = idx9 >> 4;           // fragment lane 0..31
    int ebase = idx9 & 15;           // 0 or 8
    int m     = fl & 15;
    int kb    = (fl & 16) ? 8 : 0;
    int kst   = ((fl & 16) ? 16 : 0) + ebase;
    const float4* src = (const float4*)(keyp + (size_t)(tile * 16 + m) * E_ + eb * 32 + kst);
    float4 x0 = src[0];
    float4 x1 = src[1];
    v8bf h;
    h[0] = (__bf16)x0.x; h[1] = (__bf16)x0.y; h[2] = (__bf16)x0.z; h[3] = (__bf16)x0.w;
    h[4] = (__bf16)x1.x; h[5] = (__bf16)x1.y; h[6] = (__bf16)x1.z; h[7] = (__bf16)x1.w;
    *(v8bf*)(dst + tile * 512 + fl * 16 + ebase) = h;
}

// ---------------------------------------------------------------------------
// energies[b,t] = tanh( (key@Wk)[b,t,:] + locq[b,t,:] ) . v
// block = (b, 64-row tile), 256 thr (8 waves). Wave w owns cols [16w,16w+16).
// 4 M-tiles per block reuse each B fragment; LDS A-stage is double-buffered
// (one barrier per K-step). 64 static v_wmma_f32_16x16x32_bf16 per block.
// ---------------------------------------------------------------------------
__global__ __launch_bounds__(256) void k_energy(const float* __restrict__ key,
                                                const __bf16* __restrict__ wkp,
                                                const float* __restrict__ locq,
                                                const float* __restrict__ vvec,
                                                float* __restrict__ energies) {
    __shared__ __align__(32) __bf16 aFrag[2][4 * 512];   // 8 KB double buffer
    __shared__ float eAcc[64];

    const int b    = blockIdx.x >> 4;          // T/64 = 16 tiles
    const int t0   = (blockIdx.x & 15) << 6;
    const int tid  = threadIdx.x;
    const int wave = tid >> 5;
    const int lane = tid & 31;

    const float* keyp = key + ((size_t)b * T_ + t0) * E_;

    v8f c[4] = {};

    stage_tiles(keyp, aFrag[0], 0, tid);

    for (int eb = 0; eb < E_ / 32; ++eb) {
        __syncthreads();
        if (eb + 1 < E_ / 32) stage_tiles(keyp, aFrag[(eb + 1) & 1], eb + 1, tid);

        v16bf bm = *(const v16bf*)(wkp + ((size_t)(eb * 8 + wave) * 32 + lane) * 16);
        const __bf16* abase = aFrag[eb & 1];
        #pragma unroll
        for (int mt = 0; mt < 4; ++mt) {
            v16bf a = *(const v16bf*)(abase + mt * 512 + lane * 16);
            c[mt] = __builtin_amdgcn_wmma_f32_16x16x32_bf16(false, a, false, bm,
                                                            (short)0, c[mt], false, false);
        }
    }

    // epilogue: lane holds n = 16*wave + (lane&15); VGPR r holds m = r + (lane<16?0:8)
    __syncthreads();
    if (tid < 64) eAcc[tid] = 0.f;
    __syncthreads();

    const int n    = (wave << 4) + (lane & 15);
    const int mofs = (lane & 16) ? 8 : 0;
    const float vn = vvec[n];
    #pragma unroll
    for (int mt = 0; mt < 4; ++mt) {
        #pragma unroll
        for (int r = 0; r < 8; ++r) {
            int m = mt * 16 + r + mofs;
            float x   = c[mt][r] + locq[((size_t)b * T_ + t0 + m) * A_ + n];
            float val = tanhf(x) * vn;
            val += __shfl_xor(val, 1, 32);      // sum the 16 lanes sharing m
            val += __shfl_xor(val, 2, 32);
            val += __shfl_xor(val, 4, 32);
            val += __shfl_xor(val, 8, 32);
            if ((lane & 15) == 0) atomicAdd(&eAcc[m], val);
        }
    }
    __syncthreads();
    if (tid < 64) energies[(size_t)b * T_ + t0 + tid] = eAcc[tid];
}

// ---------------------------------------------------------------------------
// masked softmax over T per batch row
// ---------------------------------------------------------------------------
__global__ __launch_bounds__(256) void k_softmax(const float* __restrict__ energies,
                                                 const unsigned char* __restrict__ mask,
                                                 float* __restrict__ weights) {
    __shared__ float red[256];
    int b = blockIdx.x, tid = threadIdx.x;
    float e[4], w[4];
    float mx = -INFINITY;
    #pragma unroll
    for (int j = 0; j < 4; ++j) {
        int t = tid + j * 256;
        float x = energies[(size_t)b * T_ + t];
        if (mask[(size_t)b * T_ + t]) x = -INFINITY;
        e[j] = x;
        mx = fmaxf(mx, x);
    }
    red[tid] = mx; __syncthreads();
    for (int s = 128; s > 0; s >>= 1) {
        if (tid < s) red[tid] = fmaxf(red[tid], red[tid + s]);
        __syncthreads();
    }
    mx = red[0]; __syncthreads();

    float sum = 0.f;
    #pragma unroll
    for (int j = 0; j < 4; ++j) { w[j] = expf(e[j] - mx); sum += w[j]; }
    red[tid] = sum; __syncthreads();
    for (int s = 128; s > 0; s >>= 1) {
        if (tid < s) red[tid] += red[tid + s];
        __syncthreads();
    }
    float inv = 1.f / red[0];
    #pragma unroll
    for (int j = 0; j < 4; ++j)
        weights[(size_t)b * T_ + tid + j * 256] = w[j] * inv;
}

// ---------------------------------------------------------------------------
// context[b,:] = sum_t w[b,t] * key[b,t,:]   (T split 8-way, f32 atomics)
// ---------------------------------------------------------------------------
__global__ __launch_bounds__(256) void k_context(const float* __restrict__ key,
                                                 const float* __restrict__ weights,
                                                 float* __restrict__ ctx) {
    __shared__ float wsh[128];
    int b   = blockIdx.x >> 3;
    int tc  = blockIdx.x & 7;
    int tid = threadIdx.x;
    if (tid < 128) wsh[tid] = weights[(size_t)b * T_ + tc * 128 + tid];
    __syncthreads();
    float acc0 = 0.f, acc1 = 0.f;
    const float* kp = key + ((size_t)b * T_ + (size_t)tc * 128) * E_;
    for (int t = 0; t < 128; ++t) {
        float w = wsh[t];
        acc0 += w * kp[(size_t)t * E_ + tid];
        acc1 += w * kp[(size_t)t * E_ + tid + 256];
    }
    atomicAdd(&ctx[(size_t)b * E_ + tid], acc0);
    atomicAdd(&ctx[(size_t)b * E_ + tid + 256], acc1);
}

// ---------------------------------------------------------------------------
extern "C" void kernel_launch(void* const* d_in, const int* in_sizes, int n_in,
                              void* d_out, int out_size, void* d_ws, size_t ws_size,
                              hipStream_t stream) {
    const float* query = (const float*)d_in[0];
    const float* key   = (const float*)d_in[1];
    const float* awcat = (const float*)d_in[2];
    const unsigned char* mask = (const unsigned char*)d_in[3];
    const float* Wq    = (const float*)d_in[4];
    const float* Wk    = (const float*)d_in[5];
    const float* convw = (const float*)d_in[6];
    const float* gamma = (const float*)d_in[7];
    const float* beta  = (const float*)d_in[8];
    const float* mean  = (const float*)d_in[9];
    const float* var   = (const float*)d_in[10];
    const float* Wloc  = (const float*)d_in[11];
    const float* v     = (const float*)d_in[12];

    float* out     = (float*)d_out;
    float* ctx     = out;                 // [B, E]  = 64*512
    float* weights = out + B_ * E_;       // [B, T]  = 64*1024

    // workspace layout (bytes)
    char* ws = (char*)d_ws;
    float*  qbuf     = (float*)(ws);                               // 32 KB
    float*  locq     = (float*)(ws + 32768);                       // 33.5 MB
    __bf16* wkp      = (__bf16*)(ws + 32768 + 33554432);           // 128 KB
    float*  energies = (float*)(ws + 32768 + 33554432 + 131072);   // 256 KB

    k_packWk <<<256, 256, 0, stream>>>(Wk, wkp);
    k_qproj  <<<B_, 128, 0, stream>>>(query, Wq, qbuf);
    k_loc    <<<B_ * 8, 128, 0, stream>>>(awcat, convw, gamma, beta, mean, var,
                                          Wloc, qbuf, locq);
    k_energy <<<B_ * (T_ / 64), 256, 0, stream>>>(key, wkp, locq, v, energies);
    k_softmax<<<B_, 256, 0, stream>>>(energies, mask, weights);
    hipMemsetAsync(ctx, 0, (size_t)B_ * E_ * sizeof(float), stream);
    k_context<<<B_ * 8, 256, 0, stream>>>(key, weights, ctx);
}
